// ConvLayer_27573690040695
// MI455X (gfx1250) — matrix operand order repaired and checked by reading, compiled, not run on hardware
//
#include <hip/hip_runtime.h>
#include <hip/hip_fp16.h>
#include <math.h>

typedef __attribute__((ext_vector_type(16))) _Float16 v16h;
typedef __attribute__((ext_vector_type(8)))  _Float16 v8h;
typedef __attribute__((ext_vector_type(8)))  float    v8f;

#define BATCH  4
#define NNODE  8192
#define MNBR   16
#define FNODE  128
#define FEDGE  64
#define KTOT   320      // 2*FNODE + FEDGE (full linear K)
#define KNE    192      // neighbor(128) + edge(64) K handled per-row
#define OTOT   256
#define NPB    16       // nodes per block
#define LDNE   200      // padded halfs per A_ne row (192+8): bank-conflict-free b128 tile reads
#define LDSELF 136      // padded halfs per A_self row (128+8)

// ---- prep: convert W (f32 [256][320] row-major) to f16 in workspace ----
__global__ __launch_bounds__(256) void convW_f16_kernel(const float* __restrict__ W,
                                                        _Float16* __restrict__ W16) {
    int i = blockIdx.x * 256 + threadIdx.x;
    if (i < OTOT * KTOT) W16[i] = (_Float16)W[i];
}

__device__ __forceinline__ float softplus_f(float x) {
    return fmaxf(x, 0.0f) + log1pf(expf(-fabsf(x)));
}

// packed f32->f16 convert + 4B LDS stores (v_cvt_pk + ds_store_b32/b64)
__device__ __forceinline__ void cvt_store8(_Float16* dst, float4 a, float4 b) {
    __half2* d2 = (__half2*)dst;
    d2[0] = __floats2half2_rn(a.x, a.y);
    d2[1] = __floats2half2_rn(a.z, a.w);
    d2[2] = __floats2half2_rn(b.x, b.y);
    d2[3] = __floats2half2_rn(b.z, b.w);
}

__device__ __forceinline__ v8f wmma_f16(v16h a, v16h b, v8f c) {
    return __builtin_amdgcn_wmma_f32_16x16x32_f16(false, a, false, b, (short)0, c, false, false);
}

// One block = 16 nodes. 256 threads = 8 waves; wave w owns output features [16w, 16w+16).
// Self-feature GEMM (K=0..127) is done ONCE per node (rows = nodes) instead of 16x,
// then the per-neighbor GEMM only runs K=192 (neighbor+edge): 0.625x the matrix FLOPs.
__global__ __launch_bounds__(256) void cgcnn_conv_wmma_kernel(
    const float*    __restrict__ node,   // [B,N,128]
    const float*    __restrict__ edge,   // [B,N,16,64]
    const int*      __restrict__ eidx,   // [B,N,16]
    const _Float16* __restrict__ W16,    // [256,320] f16
    const float*    __restrict__ bias,   // [256]
    const float*    __restrict__ alphap, // scalar
    float*          __restrict__ out)    // [B,N,128]
{
    __shared__ _Float16 AneSm[NPB * MNBR * LDNE];   // 256 rows x 192 halfs (padded) ~100KB
    __shared__ _Float16 AselfSm[NPB * LDSELF];      // 16 rows x 128 halfs (padded)
    __shared__ float    selfC_sm[NPB * OTOT];       // self contribution + bias, f32
    __shared__ int      idx_sm[NPB * MNBR];

    const int tid = threadIdx.x;
    const size_t rowBase = (size_t)blockIdx.x * (NPB * MNBR);  // first (node,m) row
    const size_t gnode0  = (size_t)blockIdx.x * NPB;           // first linear node id
    const int bb = blockIdx.x >> 9;                            // 512 blocks per batch

    // ---- stage 0: all 256 neighbor indices (one per thread) ----
    idx_sm[tid] = eidx[rowBase + tid];
    __syncthreads();

    // ---- stage 1: self features -> AselfSm (16 nodes x 128 halfs) ----
    {
        const int i  = tid >> 4;
        const int kb = (tid & 15) * 8;
        const float* src = node + (gnode0 + i) * FNODE + kb;
        float4 a = *(const float4*)(src);
        float4 b = *(const float4*)(src + 4);
        cvt_store8(&AselfSm[i * LDSELF + kb], a, b);
    }
    // ---- stage 2: gathered neighbor features -> A_ne[:, 0:128] (row = tid) ----
    {
        int j = idx_sm[tid]; if (j < 0) j = 0;      // masked later; stay in-bounds
        const float* src = node + ((size_t)bb * NNODE + (size_t)j) * FNODE;
        _Float16* dst = &AneSm[tid * LDNE];
        #pragma unroll
        for (int q = 0; q < 16; ++q) {
            float4 a = *(const float4*)(src + q * 8);
            float4 b = *(const float4*)(src + q * 8 + 4);
            cvt_store8(dst + q * 8, a, b);
        }
    }
    // ---- stage 3: edge features -> A_ne[:, 128:192] ----
    {
        const float* src = edge + (rowBase + tid) * FEDGE;
        _Float16* dst = &AneSm[tid * LDNE + FNODE];
        #pragma unroll
        for (int q = 0; q < 8; ++q) {
            float4 a = *(const float4*)(src + q * 8);
            float4 b = *(const float4*)(src + q * 8 + 4);
            cvt_store8(dst + q * 8, a, b);
        }
    }
    __syncthreads();

    const int lane = tid & 31;
    const int wid  = tid >> 5;          // 0..7
    const int l15  = lane & 15;
    const int lhi  = lane >> 4;
    const int fb   = wid * 16;          // this wave's 16 output features

    // ---- self GEMM: C_self[16 nodes x 16 feats] for filter & core halves (K=128) ----
    {
        v8f sf = {}, sc = {};
        const _Float16* bsF = W16 + (size_t)(fb +         l15) * KTOT + lhi * 16;
        const _Float16* bsC = W16 + (size_t)(fb + FNODE + l15) * KTOT + lhi * 16;
        const _Float16* aS  = &AselfSm[l15 * LDSELF + lhi * 16];
        #pragma unroll
        for (int ks = 0; ks < FNODE / 32; ++ks) {
            v8h alo = *(const v8h*)(aS + ks * 32);
            v8h ahi = *(const v8h*)(aS + ks * 32 + 8);
            v16h a = __builtin_shufflevector(alo, ahi,
                     0,1,2,3,4,5,6,7,8,9,10,11,12,13,14,15);
            sf = wmma_f16(a, *(const v16h*)(bsF + ks * 32), sf);
            sc = wmma_f16(a, *(const v16h*)(bsC + ks * 32), sc);
        }
        // C layout: lane l, vgpr r -> (node i = r + 8*lhi, feature fb + l15)
        const float bf = bias[fb + l15];
        const float bc = bias[fb + FNODE + l15];
        #pragma unroll
        for (int r = 0; r < 8; ++r) {
            selfC_sm[(r + 8 * lhi) * OTOT + fb +         l15] = sf[r] + bf;
            selfC_sm[(r + 8 * lhi) * OTOT + fb + FNODE + l15] = sc[r] + bc;
        }
    }
    __syncthreads();

    // ---- hoist all main-loop B tiles into registers: W read from L2 once per block ----
    const _Float16* bFp = W16 + (size_t)(fb +         l15) * KTOT + FNODE + lhi * 16;
    const _Float16* bCp = W16 + (size_t)(fb + FNODE + l15) * KTOT + FNODE + lhi * 16;
    v16h Bf[KNE / 32], Bc[KNE / 32];
    #pragma unroll
    for (int ks = 0; ks < KNE / 32; ++ks) {
        Bf[ks] = *(const v16h*)(bFp + ks * 32);
        Bc[ks] = *(const v16h*)(bCp + ks * 32);
    }

    const float alpha = *alphap;

    // ---- main loop: one 16(m) x 16(feat) tile pair per node, K = 192 ----
    for (int i = 0; i < NPB; ++i) {
        const _Float16* ar = &AneSm[(i * MNBR + l15) * LDNE + lhi * 16];
        v8f accF = {}, accC = {};
        #pragma unroll
        for (int ks = 0; ks < KNE / 32; ++ks) {
            v8h alo = *(const v8h*)(ar + ks * 32);
            v8h ahi = *(const v8h*)(ar + ks * 32 + 8);
            v16h a = __builtin_shufflevector(alo, ahi,
                     0,1,2,3,4,5,6,7,8,9,10,11,12,13,14,15);
            accF = wmma_f16(a, Bf[ks], accF);
            accC = wmma_f16(a, Bc[ks], accC);
        }

        // epilogue: pre-act = acc + self+bias; sum_m sigmoid(filt)*mask*softplus(core)
        const float selfF = selfC_sm[i * OTOT + fb + l15];
        const float selfC = selfC_sm[i * OTOT + fb + FNODE + l15];
        float s = 0.0f;
        #pragma unroll
        for (int r = 0; r < 8; ++r) {
            float msk = (idx_sm[i * MNBR + r + 8 * lhi] >= 0) ? 1.0f : 0.0f;
            float x = accF[r] + selfF;
            float y = accC[r] + selfC;
            float sg = 1.0f / (1.0f + expf(-x));
            s += sg * msk * softplus_f(y);
        }
        s += __shfl_xor(s, 16);                 // combine the two wave32 halves
        if (lane < 16) {
            const size_t g = gnode0 + i;
            float nv = node[g * FNODE + fb + l15];
            out[g * FNODE + fb + l15] = softplus_f(alpha * nv + s);
        }
    }
}

extern "C" void kernel_launch(void* const* d_in, const int* in_sizes, int n_in,
                              void* d_out, int out_size, void* d_ws, size_t ws_size,
                              hipStream_t stream) {
    const float*    node  = (const float*)d_in[0];
    const float*    edge  = (const float*)d_in[1];
    const int*      eidx  = (const int*)  d_in[2];
    const float*    W     = (const float*)d_in[3];
    const float*    bias  = (const float*)d_in[4];
    const float*    alpha = (const float*)d_in[5];
    float*          out   = (float*)d_out;
    _Float16*       W16   = (_Float16*)d_ws;     // 256*320*2 = 160 KB

    convW_f16_kernel<<<(OTOT * KTOT + 255) / 256, 256, 0, stream>>>(W, W16);
    cgcnn_conv_wmma_kernel<<<(BATCH * NNODE) / NPB, 256, 0, stream>>>(
        node, edge, eidx, W16, bias, alpha, out);
}
